// MessagePassingLayer_2611340116280
// MI455X (gfx1250) — matrix operand order repaired and checked by reading, not compiled, not run on hardware
//
#include <hip/hip_runtime.h>
#include <hip/hip_bf16.h>

// ---------------------------------------------------------------------------
// GNN message-passing layer for MI455X (gfx1250, wave32, WMMA).
//
// out = segment_mean( relu([x[src]|x[dst]] @ w1 + b1) @ w2 + b2, dst ) + x@ws + bs
//
// Compute-bound: 39.3 GFLOP edge MLP vs ~1 GB of (L2-resident) traffic.
// Edge MLP on V_WMMA_F32_16X16X32_BF16 (fp32 accumulate). Each wave owns a
// 32-edge tile (two 16-row M-tiles sharing every B fragment -> 2 independent
// WMMAs per DS wait). Layer-2 weight fragments are hoisted into registers.
// ---------------------------------------------------------------------------

typedef __attribute__((ext_vector_type(8)))  __bf16 v8bf;
typedef __attribute__((ext_vector_type(16))) __bf16 v16bf;
typedef __attribute__((ext_vector_type(8)))  float  v8f;

#define DIM 64          // IN_DIM == OUT_DIM == 64
#define WAVES_PER_BLOCK 8

// A-fragment (16x32 bf16, MxK) from a row-major 64-wide bf16 row.
// Lane = M + 16*g. V0-7 hold K = k0 + 16g + {0..15} (contiguous).
__device__ __forceinline__ v16bf make_a_frag(const __bf16* row, int k0, int g) {
  v8bf lo = *(const v8bf*)(row + k0 + 16 * g);
  v8bf hi = *(const v8bf*)(row + k0 + 16 * g + 8);
  v16bf a;
#pragma unroll
  for (int i = 0; i < 8; ++i) { a[i] = lo[i]; a[8 + i] = hi[i]; }
  return a;
}

// B-fragment (32x16 bf16, KxN) from LDS-resident transposed weights wT[n][k].
// Lane = N + 16*g holds K = k0 + 16g + {0..15}: one contiguous 32-byte load.
__device__ __forceinline__ v16bf make_b_frag(const __bf16* wT, int rowStride,
                                             int t, int k0, int lane) {
  int n = lane & 15, g = lane >> 4;
  return *(const v16bf*)(wT + (size_t)(t * 16 + n) * rowStride + k0 + 16 * g);
}

// ---------------------------------------------------------------------------
// Kernel 0: zero agg/deg scratch, convert x fp32 -> bf16.
// ---------------------------------------------------------------------------
__global__ void init_kernel(const float* __restrict__ x, __bf16* __restrict__ xh,
                            float* __restrict__ agg, float* __restrict__ deg,
                            int n_nodes) {
  int total = n_nodes * DIM;
  for (int i = blockIdx.x * blockDim.x + threadIdx.x; i < total;
       i += gridDim.x * blockDim.x) {
    xh[i]  = (__bf16)x[i];
    agg[i] = 0.0f;
    if (i < n_nodes) deg[i] = 0.0f;
  }
}

// ---------------------------------------------------------------------------
// Kernel 1: per-edge 2-layer MLP on WMMA + atomic mean-aggregation scatter.
// One wave per 32-edge tile: 48 v_wmma_f32_16x16x32_bf16 per tile.
// ---------------------------------------------------------------------------
__global__ void __launch_bounds__(256) edge_mlp_kernel(
    const __bf16* __restrict__ xh,
    const long long* __restrict__ ei,      // [2][E] int64
    const float* __restrict__ w1, const float* __restrict__ b1,
    const float* __restrict__ w2, const float* __restrict__ b2,
    float* __restrict__ agg, float* __restrict__ deg,
    int n_edges, int n_nodes) {
  // Transposed bf16 weights: w1T[n][k] = w1[k][n] (64x128), w2T[n][k] (64x64).
  __shared__ __align__(32) __bf16 w1T[DIM * 2 * DIM];
  __shared__ __align__(32) __bf16 w2T[DIM * DIM];
  __shared__ float b1s[DIM], b2s[DIM];
  // Per-wave staging of relu(h1) (32x64 bf16) for the C->A layout transpose.
  __shared__ __align__(32) __bf16 stage[WAVES_PER_BLOCK][32 * DIM];
  __shared__ int dstl[WAVES_PER_BLOCK][32];

  for (int idx = threadIdx.x; idx < DIM * 2 * DIM; idx += blockDim.x) {
    int n = idx >> 7, k = idx & 127;                 // row len 128
    w1T[idx] = (__bf16)w1[k * DIM + n];
  }
  for (int idx = threadIdx.x; idx < DIM * DIM; idx += blockDim.x) {
    int n = idx >> 6, k = idx & 63;                  // row len 64
    w2T[idx] = (__bf16)w2[k * DIM + n];
  }
  if (threadIdx.x < DIM) {
    b1s[threadIdx.x] = b1[threadIdx.x];
    b2s[threadIdx.x] = b2[threadIdx.x];
  }
  __syncthreads();

  const long long* srcIdx = ei;
  const long long* dstIdx = ei + n_edges;

  const int lane = threadIdx.x & 31;
  const int wave = threadIdx.x >> 5;
  const int m  = lane & 15;        // row within a 16-edge M-tile
  const int g  = lane >> 4;        // half-wave group
  const int n0 = lane & 15;        // column within a 16-wide N-tile

  // Hoist loop-invariant layer-2 B fragments into registers (8 x 8 VGPRs).
  v16bf B2r[2][4];
#pragma unroll
  for (int ks = 0; ks < 2; ++ks)
#pragma unroll
    for (int t = 0; t < 4; ++t)
      B2r[ks][t] = make_b_frag(w2T, DIM, t, ks * 32, lane);

  const int numTiles   = (n_edges + 31) >> 5;        // 32 edges per tile
  const int waveGlobal = blockIdx.x * WAVES_PER_BLOCK + wave;
  const int waveStride = gridDim.x * WAVES_PER_BLOCK;

  for (int tile = waveGlobal; tile < numTiles; tile += waveStride) {
    const long long base = (long long)tile * 32;

    // ---- gather indices for both 16-row M-tiles ----
    int rs[2], rd[2];
    bool rowValid[2];
#pragma unroll
    for (int u = 0; u < 2; ++u) {
      const long long er = base + 16 * u + m;
      rowValid[u] = er < n_edges;
      const long long ec = rowValid[u] ? er : (long long)(n_edges - 1);
      rs[u] = (int)srcIdx[ec];
      rd[u] = (int)dstIdx[ec];
      if (g == 0) {
        dstl[wave][16 * u + m] = rd[u];
        if (rowValid[u]) atomicAdd(&deg[rd[u]], 1.0f);   // degree count
      }
    }
    // Prefetch next tile's edge indices (global_prefetch_b8, speculative-safe).
    {
      const long long nb = base + (long long)waveStride * 32;
      __builtin_prefetch(srcIdx + nb + m, 0, 1);
      __builtin_prefetch(dstIdx + nb + m, 0, 1);
    }

    // ---- A fragments for concat([x[src] | x[dst]]): 2 x (16x128 bf16) ----
    v16bf A[2][4];
#pragma unroll
    for (int u = 0; u < 2; ++u) {
      const __bf16* ps = xh + (size_t)rs[u] * DIM;
      const __bf16* pd = xh + (size_t)rd[u] * DIM;
      A[u][0] = make_a_frag(ps, 0,  g);
      A[u][1] = make_a_frag(ps, 32, g);
      A[u][2] = make_a_frag(pd, 0,  g);
      A[u][3] = make_a_frag(pd, 32, g);
    }

    // ---- layer 1: (32x128) @ (128x64); B shared by both M-tiles ----
    v8f h[2][4];
#pragma unroll
    for (int u = 0; u < 2; ++u)
#pragma unroll
      for (int t = 0; t < 4; ++t) h[u][t] = (v8f){};
#pragma unroll
    for (int t = 0; t < 4; ++t) {
#pragma unroll
      for (int ks = 0; ks < 4; ++ks) {
        v16bf b = make_b_frag(w1T, 2 * DIM, t, ks * 32, lane);
        h[0][t] = __builtin_amdgcn_wmma_f32_16x16x32_bf16(
                      false, A[0][ks], false, b, (short)0, h[0][t], false, false);
        h[1][t] = __builtin_amdgcn_wmma_f32_16x16x32_bf16(
                      false, A[1][ks], false, b, (short)0, h[1][t], false, false);
      }
    }

    // ---- bias + ReLU, stage row-major 32x64 bf16 (C layout: M = r+8g+16u) ----
#pragma unroll
    for (int u = 0; u < 2; ++u)
#pragma unroll
      for (int t = 0; t < 4; ++t) {
        const float bb = b1s[t * 16 + n0];
#pragma unroll
        for (int r = 0; r < 8; ++r) {
          float v = h[u][t][r] + bb;
          v = v > 0.0f ? v : 0.0f;
          stage[wave][(16 * u + r + 8 * g) * DIM + t * 16 + n0] = (__bf16)v;
        }
      }
    // Same-wave DS RAW across lanes: drain DS queue before re-reading staging.
    asm volatile("s_wait_dscnt 0" ::: "memory");

    // ---- layer 2: (32x64) @ (64x64), B fragments already in registers ----
#pragma unroll
    for (int u = 0; u < 2; ++u) {
      const __bf16* srow = &stage[wave][(16 * u + m) * DIM];
      v16bf A2_0 = make_a_frag(srow, 0,  g);
      v16bf A2_1 = make_a_frag(srow, 32, g);
#pragma unroll
      for (int t = 0; t < 4; ++t) {
        v8f c = {};
        c = __builtin_amdgcn_wmma_f32_16x16x32_bf16(
                false, A2_0, false, B2r[0][t], (short)0, c, false, false);
        c = __builtin_amdgcn_wmma_f32_16x16x32_bf16(
                false, A2_1, false, B2r[1][t], (short)0, c, false, false);
        const float bb = b2s[t * 16 + n0];
#pragma unroll
        for (int r = 0; r < 8; ++r) {
          const int M = 16 * u + r + 8 * g;
          if (base + M < n_edges) {
            const int dn = dstl[wave][M];
            atomicAdd(&agg[(size_t)dn * DIM + t * 16 + n0], c[r] + bb);
          }
        }
      }
    }
  }
}

// ---------------------------------------------------------------------------
// Kernel 2: out = agg/clip(deg,1) + x @ ws + bs   (exact fp32 self term).
// 64 nodes per block; ws and the x tile cached in LDS.
// ---------------------------------------------------------------------------
__global__ void __launch_bounds__(256) finalize_kernel(
    const float* __restrict__ x, const float* __restrict__ wsm,
    const float* __restrict__ bsv, const float* __restrict__ agg,
    const float* __restrict__ deg, float* __restrict__ out, int n_nodes) {
  __shared__ float ws_lds[DIM * DIM];
  __shared__ float xs[64 * DIM];
  const int tid = threadIdx.x;
  const int nodeBase = blockIdx.x * 64;
  for (int i = tid; i < DIM * DIM; i += 256) ws_lds[i] = wsm[i];
  for (int i = tid; i < 64 * DIM; i += 256) {
    const int node = nodeBase + (i >> 6);
    xs[i] = (node < n_nodes) ? x[(size_t)node * DIM + (i & 63)] : 0.0f;
  }
  __syncthreads();

  const int j   = tid & 63;
  const int grp = tid >> 6;            // 4 groups x 16 nodes
  const float bsj = bsv[j];
  for (int nl = grp; nl < 64; nl += 4) {
    const int node = nodeBase + nl;
    if (node >= n_nodes) break;
    float d = deg[node];
    d = d < 1.0f ? 1.0f : d;
    float acc = agg[(size_t)node * DIM + j] / d + bsj;
#pragma unroll
    for (int k = 0; k < DIM; ++k)
      acc = fmaf(xs[nl * DIM + k], ws_lds[k * DIM + j], acc);
    out[(size_t)node * DIM + j] = acc;
  }
}

// ---------------------------------------------------------------------------
extern "C" void kernel_launch(void* const* d_in, const int* in_sizes, int n_in,
                              void* d_out, int out_size, void* d_ws, size_t ws_size,
                              hipStream_t stream) {
  const float*     x   = (const float*)d_in[0];
  const long long* ei  = (const long long*)d_in[1];
  const float*     w1  = (const float*)d_in[2];
  const float*     b1  = (const float*)d_in[3];
  const float*     w2  = (const float*)d_in[4];
  const float*     b2  = (const float*)d_in[5];
  const float*     wsm = (const float*)d_in[6];
  const float*     bs  = (const float*)d_in[7];

  const int n_nodes = in_sizes[0] / DIM;
  const int n_edges = in_sizes[1] / 2;

  // Workspace carve-out: xh (bf16 N x 64), agg (f32 N x 64), deg (f32 N).
  char* p = (char*)d_ws;
  __bf16* xh = (__bf16*)p;
  p += (((size_t)n_nodes * DIM * 2) + 255) & ~(size_t)255;
  float* agg = (float*)p;
  p += (((size_t)n_nodes * DIM * 4) + 255) & ~(size_t)255;
  float* deg = (float*)p;

  init_kernel<<<1024, 256, 0, stream>>>(x, xh, agg, deg, n_nodes);

  edge_mlp_kernel<<<608, 256, 0, stream>>>(xh, ei, w1, b1, w2, b2, agg, deg,
                                           n_edges, n_nodes);

  finalize_kernel<<<(n_nodes + 63) / 64, 256, 0, stream>>>(
      x, wsm, bs, agg, deg, (float*)d_out, n_nodes);
}
